// Memory_1623497638569
// MI455X (gfx1250) — matrix operand order repaired and checked by reading, not compiled, not run on hardware
//
#include <hip/hip_runtime.h>
#include <cstdint>

typedef __attribute__((ext_vector_type(16))) _Float16 v16h;
typedef __attribute__((ext_vector_type(8)))  float    v8f;

#define B_ 256
#define C_ 1024
#define D_ 512
#define M_ 4096

union Frag { v16h h; uint32_t u[8]; unsigned short us[16]; };

// ---------- K1: mean over C -> attn buffer [B, D] ----------
__global__ __launch_bounds__(256) void k_mean_c(const float* __restrict__ f,
                                                float* __restrict__ attn) {
  int b  = blockIdx.x >> 1;
  int dt = ((blockIdx.x & 1) << 8) + threadIdx.x;
  const float* p = f + (size_t)b * C_ * D_ + dt;
  float s = 0.0f;
  for (int c = 0; c < C_; ++c) {
    __builtin_prefetch(p + (size_t)(c + 16) * D_, 0, 0);   // global_prefetch_b8
    s += p[(size_t)c * D_];
  }
  attn[b * D_ + dt] = s * (1.0f / C_);
}

// ---------- K2: softmax over D (in place) ----------
__global__ __launch_bounds__(256) void k_softmax_d(float* __restrict__ attn) {
  int b = blockIdx.x, t = threadIdx.x;
  __shared__ float s[256];
  float v0 = attn[b * D_ + t], v1 = attn[b * D_ + t + 256];
  s[t] = fmaxf(v0, v1); __syncthreads();
  for (int st = 128; st > 0; st >>= 1) { if (t < st) s[t] = fmaxf(s[t], s[t + st]); __syncthreads(); }
  float mx = s[0]; __syncthreads();
  float e0 = __expf(v0 - mx), e1 = __expf(v1 - mx);
  s[t] = e0 + e1; __syncthreads();
  for (int st = 128; st > 0; st >>= 1) { if (t < st) s[t] += s[t + st]; __syncthreads(); }
  float inv = 1.0f / s[0];
  attn[b * D_ + t] = e0 * inv;
  attn[b * D_ + t + 256] = e1 * inv;
}

// ---------- K3: feature_G[b,c] = (1/D) dot(feature[b,c,:], attn[b,:]) ----------
__global__ __launch_bounds__(256) void k_feat_g(const float* __restrict__ f,
                                                const float* __restrict__ attn,
                                                float* __restrict__ featG,
                                                _Float16* __restrict__ featG_h) {
  __shared__ float sa[D_];
  int b = blockIdx.x >> 7;               // 128 blocks per batch row
  int t = threadIdx.x;
  sa[t] = attn[b * D_ + t];
  sa[t + 256] = attn[b * D_ + t + 256];
  __syncthreads();
  int w = t >> 5, lane = t & 31;
  int c = ((blockIdx.x & 127) << 3) + w; // wave per (b,c) row
  const float* row = f + ((size_t)b * C_ + c) * D_;
  float s = 0.0f;
  for (int i = lane; i < D_; i += 32) s += row[i] * sa[i];
  for (int o = 16; o > 0; o >>= 1) s += __shfl_xor(s, o, 32);
  if (lane == 0) {
    float v = s * (1.0f / D_);
    featG[b * C_ + c] = v;
    featG_h[b * C_ + c] = (_Float16)v;
  }
}

// ---------- K4: memory f32 -> f16 (row-major copy) + f16 transpose [C, M] ----------
__global__ __launch_bounds__(256) void k_cvt_transpose(const float* __restrict__ m,
                                                       _Float16* __restrict__ mh,
                                                       _Float16* __restrict__ mhT) {
  __shared__ float tile[32][33];
  int m0 = (blockIdx.x >> 5) << 5;   // 128 M-tiles
  int c0 = (blockIdx.x & 31) << 5;   // 32 C-tiles
  int tc = threadIdx.x & 31;
  int tr = threadIdx.x >> 5;         // 0..7
#pragma unroll
  for (int r = 0; r < 4; ++r) {
    int row = tr + (r << 3);         // 0..31
    float v = m[(size_t)(m0 + row) * C_ + c0 + tc];
    mh[(size_t)(m0 + row) * C_ + c0 + tc] = (_Float16)v;
    tile[row][tc] = v;
  }
  __syncthreads();
#pragma unroll
  for (int r = 0; r < 4; ++r) {
    int row = tr + (r << 3);         // c-index within tile
    mhT[(size_t)(c0 + row) * M_ + m0 + tc] = (_Float16)tile[tc][row];
  }
}

// ---------- K5: WMMA GEMM1  score[B,M] = featG_h @ mem_h^T ----------
__global__ __launch_bounds__(256) void k_gemm_score(const _Float16* __restrict__ A,
                                                    const _Float16* __restrict__ Bm,
                                                    float* __restrict__ S) {
  const int wave = blockIdx.x * 8 + (threadIdx.x >> 5);
  const int lane = threadIdx.x & 31;
  const int lm = lane & 15, hi = lane >> 4;
  const int tm = wave >> 8;          // B/16 = 16 row tiles
  const int tn = wave & 255;         // M/16 = 256 col tiles
  const _Float16* arow = A  + (size_t)(tm * 16 + lm) * C_;
  const _Float16* brow = Bm + (size_t)(tn * 16 + lm) * C_; // memory row = N index, contiguous K
  const int kha = hi * 8, khb = hi * 16;
  v8f acc = {};
  for (int kb = 0; kb < C_; kb += 32) {
    Frag a, b;
#pragma unroll
    for (int j = 0; j < 8; ++j) {
      int koff = (j < 4) ? (2 * j + kha) : (16 + 2 * (j - 4) + kha);
      a.u[j] = *reinterpret_cast<const uint32_t*>(arow + kb + koff);
      b.u[j] = *reinterpret_cast<const uint32_t*>(brow + kb + koff);
    }
    acc = __builtin_amdgcn_wmma_f32_16x16x32_f16(false, a.h, false, b.h,
                                                 (short)0, acc, false, false);
  }
#pragma unroll
  for (int r = 0; r < 8; ++r) {
    int row = tm * 16 + r + hi * 8;
    int col = tn * 16 + lm;
    S[(size_t)row * M_ + col] = acc[r];
  }
}

// ---------- K6: row softmax over M (score_image -> f16) + argmax ----------
__global__ __launch_bounds__(256) void k_softmax_m(const float* __restrict__ score,
                                                   _Float16* __restrict__ P,
                                                   int* __restrict__ idxOut) {
  int b = blockIdx.x, t = threadIdx.x;
  __shared__ float sv[256];
  __shared__ int   si[256];
  const float* row = score + (size_t)b * M_;
  float vals[16];
  float mx = -INFINITY; int mi = 0;
#pragma unroll
  for (int k = 0; k < 16; ++k) {
    int j = t + (k << 8);
    float v = row[j]; vals[k] = v;
    if (v > mx) { mx = v; mi = j; }
  }
  sv[t] = mx; si[t] = mi; __syncthreads();
  for (int st = 128; st > 0; st >>= 1) {
    if (t < st) {
      float o = sv[t + st];
      if (o > sv[t] || (o == sv[t] && si[t + st] < si[t])) { sv[t] = o; si[t] = si[t + st]; }
    }
    __syncthreads();
  }
  float rmx = sv[0]; int rix = si[0]; __syncthreads();
  float ps = 0.0f;
#pragma unroll
  for (int k = 0; k < 16; ++k) { vals[k] = __expf(vals[k] - rmx); ps += vals[k]; }
  sv[t] = ps; __syncthreads();
  for (int st = 128; st > 0; st >>= 1) { if (t < st) sv[t] += sv[t + st]; __syncthreads(); }
  float inv = 1.0f / sv[0];
#pragma unroll
  for (int k = 0; k < 16; ++k)
    P[(size_t)b * M_ + t + (k << 8)] = (_Float16)(vals[k] * inv);
  if (t == 0) idxOut[b] = rix;
}

// ---------- K7: weights[b] = softmax over batch of column idx[b], element b ----------
__global__ __launch_bounds__(256) void k_col_weight(const float* __restrict__ score,
                                                    const int* __restrict__ idx,
                                                    float* __restrict__ weights) {
  int b = blockIdx.x, t = threadIdx.x;
  __shared__ float s[256];
  int j = idx[b];
  float v = score[(size_t)t * M_ + j];
  s[t] = v; __syncthreads();
  for (int st = 128; st > 0; st >>= 1) { if (t < st) s[t] = fmaxf(s[t], s[t + st]); __syncthreads(); }
  float mx = s[0]; __syncthreads();
  float e = __expf(v - mx);
  s[t] = e; __syncthreads();
  for (int st = 128; st > 0; st >>= 1) { if (t < st) s[t] += s[t + st]; __syncthreads(); }
  if (t == b) weights[b] = e / s[0];
}

// ---------- K8: WMMA GEMM2  mr[B,C] = P_h @ memory + featG  (B from mem_hT) ----------
__global__ __launch_bounds__(256) void k_gemm_mr(const _Float16* __restrict__ P,
                                                 const _Float16* __restrict__ BmT,
                                                 const float* __restrict__ featG,
                                                 float* __restrict__ MR) {
  const int wave = blockIdx.x * 8 + (threadIdx.x >> 5);
  const int lane = threadIdx.x & 31;
  const int lm = lane & 15, hi = lane >> 4;
  const int tm = wave >> 6;          // B/16 = 16 row tiles
  const int tn = wave & 63;          // C/16 = 64 col tiles
  const _Float16* arow = P   + (size_t)(tm * 16 + lm) * M_;
  const _Float16* brow = BmT + (size_t)(tn * 16 + lm) * M_; // mhT[n, k] contiguous in K
  const int kha = hi * 8, khb = hi * 16;
  v8f acc = {};
  for (int kb = 0; kb < M_; kb += 32) {
    Frag a, b;
#pragma unroll
    for (int j = 0; j < 8; ++j) {
      int koff = (j < 4) ? (2 * j + kha) : (16 + 2 * (j - 4) + kha);
      a.u[j] = *reinterpret_cast<const uint32_t*>(arow + kb + koff);
      b.u[j] = *reinterpret_cast<const uint32_t*>(brow + kb + koff);
    }
    acc = __builtin_amdgcn_wmma_f32_16x16x32_f16(false, a.h, false, b.h,
                                                 (short)0, acc, false, false);
  }
#pragma unroll
  for (int r = 0; r < 8; ++r) {
    int row = tm * 16 + r + hi * 8;
    int col = tn * 16 + lm;
    MR[(size_t)row * C_ + col] = acc[r] + featG[(size_t)row * C_ + col];
  }
}

// ---------- K9: global_compensation = feature + mr[b,c]  (float4 streaming) ----------
__global__ __launch_bounds__(256) void k_final_add(const float4* __restrict__ f,
                                                   const float* __restrict__ mr,
                                                   float4* __restrict__ out) {
  size_t i = (size_t)blockIdx.x * 256 + threadIdx.x;  // float4 index
  int bc = (int)(i >> 7);                             // D/4 = 128 vec4 per (b,c)
  float m = mr[bc];
  float4 v = f[i];
  v.x += m; v.y += m; v.z += m; v.w += m;
  out[i] = v;
}

// ---------- K10: updated_memory row: memory + scatter(update) then normalize ----------
__global__ __launch_bounds__(256) void k_update_mem(const float* __restrict__ memory,
                                                    const float* __restrict__ featG,
                                                    const int* __restrict__ idx,
                                                    const float* __restrict__ weights,
                                                    const uint8_t* __restrict__ mask,
                                                    float* __restrict__ out) {
  int m = blockIdx.x, t = threadIdx.x;
  __shared__ float sw[256];
  __shared__ int   si[256];
  __shared__ float s[256];
  sw[t] = weights[t] * (mask[t] ? 1.0f : 0.0f);
  si[t] = idx[t];
  __syncthreads();
  float acc[4];
#pragma unroll
  for (int k = 0; k < 4; ++k) acc[k] = memory[(size_t)m * C_ + t + (k << 8)];
  for (int b = 0; b < B_; ++b) {
    if (si[b] == m) {                   // uniform branch across block
      float w = sw[b];
#pragma unroll
      for (int k = 0; k < 4; ++k) acc[k] += featG[(size_t)b * C_ + t + (k << 8)] * w;
    }
  }
  float ss = 0.0f;
#pragma unroll
  for (int k = 0; k < 4; ++k) ss += acc[k] * acc[k];
  s[t] = ss; __syncthreads();
  for (int st = 128; st > 0; st >>= 1) { if (t < st) s[t] += s[t + st]; __syncthreads(); }
  float inv = 1.0f / fmaxf(sqrtf(s[0]), 1e-12f);
#pragma unroll
  for (int k = 0; k < 4; ++k) out[(size_t)m * C_ + t + (k << 8)] = acc[k] * inv;
}

extern "C" void kernel_launch(void* const* d_in, const int* in_sizes, int n_in,
                              void* d_out, int out_size, void* d_ws, size_t ws_size,
                              hipStream_t stream) {
  const float*   feature = (const float*)d_in[0];
  const float*   memory  = (const float*)d_in[1];
  const uint8_t* mask    = (const uint8_t*)d_in[2];
  (void)in_sizes; (void)n_in; (void)out_size; (void)ws_size;

  float* out_gc = (float*)d_out;
  float* out_um = out_gc + (size_t)B_ * C_ * D_;

  char* ws = (char*)d_ws;
  float*    attn    = (float*)(ws + 0);                 // 512 KB
  float*    featG   = (float*)(ws + 524288);            // 1 MB
  _Float16* featG_h = (_Float16*)(ws + 1572864);        // 512 KB
  _Float16* mem_h   = (_Float16*)(ws + 2097152);        // 8 MB   [M, C]
  _Float16* mem_hT  = (_Float16*)(ws + 10485760);       // 8 MB   [C, M]
  float*    score   = (float*)(ws + 18874368);          // 4 MB
  _Float16* P_h     = (_Float16*)(ws + 23068672);       // 2 MB
  int*      idx     = (int*)(ws + 25165824);            // 1 KB
  float*    weights = (float*)(ws + 25166848);          // 1 KB
  float*    mr      = (float*)(ws + 25167872);          // 1 MB

  k_mean_c        <<<B_ * 2,           256, 0, stream>>>(feature, attn);
  k_softmax_d     <<<B_,               256, 0, stream>>>(attn);
  k_feat_g        <<<(B_ * C_) / 8,    256, 0, stream>>>(feature, attn, featG, featG_h);
  k_cvt_transpose <<<(M_ / 32) * (C_ / 32), 256, 0, stream>>>(memory, mem_h, mem_hT);
  k_gemm_score    <<<(16 * 256) / 8,   256, 0, stream>>>(featG_h, mem_h, score);
  k_softmax_m     <<<B_,               256, 0, stream>>>(score, P_h, idx);
  k_col_weight    <<<B_,               256, 0, stream>>>(score, idx, weights);
  k_gemm_mr       <<<(16 * 64) / 8,    256, 0, stream>>>(P_h, mem_hT, featG, mr);
  k_final_add     <<<(B_ * C_ * D_) / 4 / 256, 256, 0, stream>>>((const float4*)feature, mr, (float4*)out_gc);
  k_update_mem    <<<M_,               256, 0, stream>>>(memory, featG, idx, weights, mask, out_um);
}
